// MultiAttentionHeads_18408229830861
// MI455X (gfx1250) — matrix operand order repaired and checked by reading, compiled, not run on hardware
//
#include <hip/hip_runtime.h>

typedef __attribute__((ext_vector_type(16))) _Float16 v16h;
typedef __attribute__((ext_vector_type(8)))  _Float16 v8h;
typedef __attribute__((ext_vector_type(4)))  _Float16 v4h;
typedef __attribute__((ext_vector_type(8)))  float    v8f;

#define DIM    1024
#define NHEAD  16
#define HDIM   64
#define SEQ    2048
#define BATCH  4
#define NROWS  (BATCH * SEQ)   // 8192

#define USE_ASYNC 1   // gfx1250 GLOBAL_LOAD_ASYNC_TO_LDS_B128 staging path

// ---------------------------------------------------------------------------
// CDNA5 async global->LDS copy (16B per lane), ASYNCcnt-tracked.
// Generic LDS pointer low 32 bits == wave-relative LDS byte offset
// (addrspacecast(3->generic) = {aperture_hi, lds_offset}; ISA §10.2).
// ---------------------------------------------------------------------------
__device__ __forceinline__ void async_copy_b128(void* lds_dst, const void* gsrc) {
#if USE_ASYNC
  asm volatile("global_load_async_to_lds_b128 %0, %1, off"
               :
               : "v"((unsigned)(size_t)lds_dst),
                 "v"((unsigned long long)(size_t)gsrc)
               : "memory");
#else
  *(v8h*)lds_dst = *(const v8h*)gsrc;
#endif
}

__device__ __forceinline__ void async_join() {
#if USE_ASYNC
#if __has_builtin(__builtin_amdgcn_s_wait_asynccnt)
  __builtin_amdgcn_s_wait_asynccnt(0);
#else
  asm volatile("s_wait_asynccnt 0x0" ::: "memory");
#endif
#endif
}

// ---------------------------------------------------------------------------
// WMMA helpers (CDNA5 wave32, 16x16x32 f16 -> f32)
// ---------------------------------------------------------------------------
__device__ __forceinline__ v8f wmma_f16f32(v16h a, v16h b, v8f c) {
  return __builtin_amdgcn_wmma_f32_16x16x32_f16(
      /*neg_a=*/false, a, /*neg_b=*/false, b,
      /*c_mod=*/(short)0, c, /*reuse_a=*/false, /*reuse_b=*/false);
}

// A-matrix 16x32 fragment. src points at (row0, k0); ld = row stride (halves).
// Lanes 0-15: row=lane, K 0..7 & 16..23; lanes 16-31: K 8..15 & 24..31.
__device__ __forceinline__ v16h load_a_frag(const _Float16* src, int ld, int lane) {
  const int l16  = lane & 15;
  const int koff = (lane & 16) ? 8 : 0;
  const _Float16* p = src + l16 * ld + koff;
  v8h lo = *(const v8h*)(p);
  v8h hi = *(const v8h*)(p + 16);
  v16h r;
#pragma unroll
  for (int i = 0; i < 8; ++i) { r[i] = lo[i]; r[i + 8] = hi[i]; }
  return r;
}

// B-matrix 32x16 fragment from transposed storage: (k, col) at srcT[col*ld+k].
// Lanes 0-15: col=lane, K 0..7 & 16..23; lanes 16-31: K 8..15 & 24..31.
__device__ __forceinline__ v16h load_b_frag(const _Float16* srcT, int ld, int lane) {
  const int l16  = lane & 15;
  const int koff = (lane & 16) ? 8 : 0;
  const _Float16* p = srcT + l16 * ld + koff;
  v8h lo = *(const v8h*)(p);
  v8h hi = *(const v8h*)(p + 16);
  v16h r;
#pragma unroll
  for (int i = 0; i < 8; ++i) { r[i] = lo[i]; r[i + 8] = hi[i]; }
  return r;
}

// ---------------------------------------------------------------------------
// f32 -> f16 down-convert (activations)
// ---------------------------------------------------------------------------
__global__ __launch_bounds__(256) void cvt_f32_f16(const float4* __restrict__ in,
                                                   v4h* __restrict__ out, int n4) {
  const int i = blockIdx.x * 256 + threadIdx.x;
  if (i < n4) {
    float4 x = in[i];
    v4h y;
    y[0] = (_Float16)x.x; y[1] = (_Float16)x.y;
    y[2] = (_Float16)x.z; y[3] = (_Float16)x.w;
    out[i] = y;
  }
}

// ---------------------------------------------------------------------------
// Weight convert + transpose: W[k][n] f32 -> Wt[n][k] f16 (64x64 LDS tiles).
// ---------------------------------------------------------------------------
__global__ __launch_bounds__(256) void cvt_w_transpose(const float* __restrict__ W,
                                                       _Float16* __restrict__ Wt) {
  __shared__ _Float16 T[64][72];
  const int k0 = blockIdx.y * 64;
  const int n0 = blockIdx.x * 64;
  const int r  = threadIdx.x >> 2;          // 0..63
  const int sg = (threadIdx.x & 3) * 16;    // 0/16/32/48

  const float4* src = (const float4*)(W + (size_t)(k0 + r) * DIM + n0 + sg);
#pragma unroll
  for (int q = 0; q < 4; ++q) {
    float4 x = src[q];
    T[r][sg + q * 4 + 0] = (_Float16)x.x;
    T[r][sg + q * 4 + 1] = (_Float16)x.y;
    T[r][sg + q * 4 + 2] = (_Float16)x.z;
    T[r][sg + q * 4 + 3] = (_Float16)x.w;
  }
  __syncthreads();
  v8h o0, o1;
#pragma unroll
  for (int j = 0; j < 8; ++j) { o0[j] = T[sg + j][r]; o1[j] = T[sg + 8 + j][r]; }
  _Float16* dst = Wt + (size_t)(n0 + r) * DIM + k0 + sg;
  *(v8h*)(dst)     = o0;
  *(v8h*)(dst + 8) = o1;
}

// ---------------------------------------------------------------------------
// Tiled WMMA GEMM:  C[M,N] = A[M,K] @ Wt^T + bias  (Wt is N x K, k-contiguous)
// Double-buffered async global->LDS staging: tile t+1 streams in while tile t
// is consumed by WMMAs. One barrier per K-step.
// ---------------------------------------------------------------------------
template <typename OutT, bool ROPE, bool VTRANS>
__global__ __launch_bounds__(256) void gemm_wmma(const _Float16* __restrict__ A,
                                                 const _Float16* __restrict__ Wt,
                                                 const float* __restrict__ bias,
                                                 OutT* __restrict__ C,
                                                 int K, int N) {
  constexpr int BM = 128, BN = 128, BK = 32;
  constexpr int LDT = BK + 8;  // 40-half padded stride (80B)
  __shared__ _Float16 As[2][BM * LDT];   // [row][k]
  __shared__ _Float16 Bs[2][BN * LDT];   // [col][k]

  const int tid  = threadIdx.x;
  const int lane = tid & 31;
  const int wave = tid >> 5;
  const int wm   = wave >> 2;  // 0..1
  const int wn   = wave & 3;   // 0..3
  const int m0   = blockIdx.y * BM;
  const int n0   = blockIdx.x * BN;

  const v8f vzero = {};
  v8f acc[4][2];
#pragma unroll
  for (int i = 0; i < 4; ++i)
#pragma unroll
    for (int j = 0; j < 2; ++j) acc[i][j] = vzero;

  const int srow = tid >> 1;           // 0..127
  const int sseg = (tid & 1) * 16;     // 0 / 16
  const _Float16* aG = A  + (size_t)(m0 + srow) * K + sseg;
  const _Float16* bG = Wt + (size_t)(n0 + srow) * K + sseg;
  const int lOff = srow * LDT + sseg;

  auto issue_tile = [&](int kt, int buf) {
    async_copy_b128(&As[buf][lOff],     aG + kt);
    async_copy_b128(&As[buf][lOff] + 8, aG + kt + 8);
    async_copy_b128(&Bs[buf][lOff],     bG + kt);
    async_copy_b128(&Bs[buf][lOff] + 8, bG + kt + 8);
  };

  issue_tile(0, 0);
  const int nkt = K / BK;
  for (int t = 0; t < nkt; ++t) {
    const int buf = t & 1;
    async_join();        // this wave's copies for tile t have landed
    __syncthreads();     // all waves' copies landed; all done reading buf^1
    if (t + 1 < nkt) issue_tile((t + 1) * BK, buf ^ 1);

    v16h af[4], bf[2];
#pragma unroll
    for (int i = 0; i < 4; ++i)
      af[i] = load_a_frag(&As[buf][(wm * 64 + i * 16) * LDT], LDT, lane);
#pragma unroll
    for (int j = 0; j < 2; ++j)
      bf[j] = load_b_frag(&Bs[buf][(wn * 32 + j * 16) * LDT], LDT, lane);
#pragma unroll
    for (int i = 0; i < 4; ++i)
#pragma unroll
      for (int j = 0; j < 2; ++j)
        acc[i][j] = wmma_f16f32(af[i], bf[j], acc[i][j]);
  }

  // ---- epilogue: bias (+ fused RoPE) ----
  const int l16   = lane & 15;
  const int half8 = (lane & 16) ? 8 : 0;
#pragma unroll
  for (int i = 0; i < 4; ++i) {
#pragma unroll
    for (int j = 0; j < 2; ++j) {
      const int col  = n0 + wn * 32 + j * 16 + l16;
      const float bc = bias[col];
#pragma unroll
      for (int v = 0; v < 8; ++v) {
        const int row = m0 + wm * 64 + i * 16 + v + half8;
        float x = acc[i][j][v] + bc;
        if constexpr (ROPE) {
          const int   t    = row & (SEQ - 1);
          const int   d    = col & (HDIM - 1);
          const float invf = __expf(-0.2878231366242557f * (float)(d >> 1)); // ln(1e4)/32
          float sn, cs;
          __sincosf((float)t * invf, &sn, &cs);
          const float prt = __shfl_xor(x, 1, 32);   // even<->odd RoPE pair
          x = x * cs + ((d & 1) ? prt : -prt) * sn;
        }
        if constexpr (VTRANS) {
          // [b][h*64+d][s]: b = row>>11, s = row&2047
          const size_t idx = ((size_t)(row >> 11) << 21) + (size_t)col * SEQ + (row & (SEQ - 1));
          C[idx] = (OutT)x;
        } else {
          C[(size_t)row * N + col] = (OutT)x;
        }
      }
    }
  }
}

// ---------------------------------------------------------------------------
// Flash attention fwd (causal). Block = 128 query rows (8 waves x 16 rows).
// Double-buffered async K/V tile streaming; online softmax in f32.
// Grid: (SEQ/128, BATCH*NHEAD).
// ---------------------------------------------------------------------------
__global__ __launch_bounds__(256) void attn_fwd(const _Float16* __restrict__ Q,
                                                const _Float16* __restrict__ Km,
                                                const _Float16* __restrict__ Vt_g,
                                                _Float16* __restrict__ O) {
  constexpr int LDS_S = HDIM + 8;  // 72-half stride (144B)
  __shared__ _Float16 Ks[2][64 * LDS_S];    // [s][d]
  __shared__ _Float16 Vt[2][64 * LDS_S];    // [d][s]
  __shared__ _Float16 Ps[8][16 * LDS_S];    // per-wave P re-layout slab

  const int tid   = threadIdx.x;
  const int lane  = tid & 31;
  const int wave  = tid >> 5;
  const int l16   = lane & 15;
  const int half8 = (lane & 16) ? 8 : 0;

  const int bh = blockIdx.y;
  const int b  = bh >> 4;
  const int h  = bh & 15;
  const int t0 = blockIdx.x * 128;
  const int tw = t0 + wave * 16;

  const _Float16* qbase = Q + ((size_t)(b * SEQ + tw)) * DIM + h * HDIM;
  const v16h aq0 = load_a_frag(qbase, DIM, lane);
  const v16h aq1 = load_a_frag(qbase + 32, DIM, lane);

  const v8f vzero = {};
  float m_st[8], l_st[8];
  v8f oacc[4];
#pragma unroll
  for (int v = 0; v < 8; ++v) { m_st[v] = -1e30f; l_st[v] = 0.0f; }
#pragma unroll
  for (int j = 0; j < 4; ++j) oacc[j] = vzero;

  const int ntiles = (t0 >> 6) + 2;     // causal tile bound
  const int srow   = tid >> 2;          // 0..63
  const int sseg   = (tid & 3) * 16;    // 0/16/32/48

  const _Float16* kG = Km   + ((size_t)(b * SEQ + srow)) * DIM + h * HDIM + sseg;
  const _Float16* vG = Vt_g + ((size_t)(bh * HDIM + srow)) * SEQ + sseg;
  const int lOff = srow * LDS_S + sseg;

  auto issue_kv = [&](int s0, int buf) {
    async_copy_b128(&Ks[buf][lOff],     kG + (size_t)s0 * DIM);   // K rows: s-major
    async_copy_b128(&Ks[buf][lOff] + 8, kG + (size_t)s0 * DIM + 8);
    async_copy_b128(&Vt[buf][lOff],     vG + s0);                 // V rows: d-major
    async_copy_b128(&Vt[buf][lOff] + 8, vG + s0 + 8);
  };

  issue_kv(0, 0);
  for (int st = 0; st < ntiles; ++st) {
    const int s0  = st * 64;
    const int buf = st & 1;
    async_join();
    __syncthreads();
    if (st + 1 < ntiles) issue_kv(s0 + 64, buf ^ 1);

    // ---- scores: S = (Q @ K^T) * 1/sqrt(D) ----
    v8f sc[4];
#pragma unroll
    for (int jn = 0; jn < 4; ++jn) {
      v16h bk0 = load_b_frag(&Ks[buf][(jn * 16) * LDS_S + 0], LDS_S, lane);
      v16h bk1 = load_b_frag(&Ks[buf][(jn * 16) * LDS_S + 32], LDS_S, lane);
      v8f z = vzero;
      z = wmma_f16f32(aq0, bk0, z);
      z = wmma_f16f32(aq1, bk1, z);
      sc[jn] = z;
    }
    // ---- scale + causal mask ----
#pragma unroll
    for (int jn = 0; jn < 4; ++jn) {
      const int scol = s0 + jn * 16 + l16;
#pragma unroll
      for (int v = 0; v < 8; ++v) {
        const int trow = tw + v + half8;
        const float x  = sc[jn][v] * 0.125f;
        sc[jn][v] = (scol > trow) ? -1e30f : x;
      }
    }
    // ---- online softmax ----
    float resc[8], rs[8];
#pragma unroll
    for (int v = 0; v < 8; ++v) {
      float r = fmaxf(fmaxf(sc[0][v], sc[1][v]), fmaxf(sc[2][v], sc[3][v]));
      r = fmaxf(r, __shfl_xor(r, 8, 32));
      r = fmaxf(r, __shfl_xor(r, 4, 32));
      r = fmaxf(r, __shfl_xor(r, 2, 32));
      r = fmaxf(r, __shfl_xor(r, 1, 32));
      const float mn = fmaxf(m_st[v], r);
      resc[v] = __expf(m_st[v] - mn);
      m_st[v] = mn;
      rs[v]   = 0.0f;
    }
#pragma unroll
    for (int jn = 0; jn < 4; ++jn) {
#pragma unroll
      for (int v = 0; v < 8; ++v) {
        const float p = __expf(sc[jn][v] - m_st[v]);
        rs[v] += p;
        Ps[wave][(v + half8) * LDS_S + jn * 16 + l16] = (_Float16)p;
      }
    }
#pragma unroll
    for (int v = 0; v < 8; ++v) {
      float r = rs[v];
      r += __shfl_xor(r, 8, 32);
      r += __shfl_xor(r, 4, 32);
      r += __shfl_xor(r, 2, 32);
      r += __shfl_xor(r, 1, 32);
      l_st[v] = l_st[v] * resc[v] + r;
    }
#pragma unroll
    for (int jd = 0; jd < 4; ++jd)
#pragma unroll
      for (int v = 0; v < 8; ++v) oacc[jd][v] *= resc[v];

    // ---- O += P @ V ----
    const v16h ap0 = load_a_frag(&Ps[wave][0], LDS_S, lane);
    const v16h ap1 = load_a_frag(&Ps[wave][32], LDS_S, lane);
#pragma unroll
    for (int jd = 0; jd < 4; ++jd) {
      v16h bv0 = load_b_frag(&Vt[buf][(jd * 16) * LDS_S + 0], LDS_S, lane);
      v16h bv1 = load_b_frag(&Vt[buf][(jd * 16) * LDS_S + 32], LDS_S, lane);
      oacc[jd] = wmma_f16f32(ap0, bv0, oacc[jd]);
      oacc[jd] = wmma_f16f32(ap1, bv1, oacc[jd]);
    }
  }

  // ---- normalize and write f16 staging for the output GEMM ----
#pragma unroll
  for (int jd = 0; jd < 4; ++jd) {
#pragma unroll
    for (int v = 0; v < 8; ++v) {
      const int t   = tw + v + half8;
      const int col = h * HDIM + jd * 16 + l16;
      O[((size_t)(b * SEQ + t)) * DIM + col] = (_Float16)(oacc[jd][v] / l_st[v]);
    }
  }
}

// ---------------------------------------------------------------------------
// Host-side orchestration
// ---------------------------------------------------------------------------
extern "C" void kernel_launch(void* const* d_in, const int* in_sizes, int n_in,
                              void* d_out, int out_size, void* d_ws, size_t ws_size,
                              hipStream_t stream) {
  const float* query = (const float*)d_in[0];
  const float* key   = (const float*)d_in[1];
  const float* value = (const float*)d_in[2];
  const float* Wq    = (const float*)d_in[3];
  const float* bq    = (const float*)d_in[4];
  const float* Wk    = (const float*)d_in[5];
  const float* bk    = (const float*)d_in[6];
  const float* Wv    = (const float*)d_in[7];
  const float* bv    = (const float*)d_in[8];
  const float* Wo    = (const float*)d_in[9];
  const float* bo    = (const float*)d_in[10];
  float* out = (float*)d_out;

  const size_t nAct = (size_t)NROWS * DIM;   // 8,388,608
  const size_t nW   = (size_t)DIM * DIM;     // 1,048,576

  char* p = (char*)d_ws;
  _Float16* qf   = (_Float16*)p; p += nAct * 2;   // reused as attention output
  _Float16* kf   = (_Float16*)p; p += nAct * 2;
  _Float16* vf   = (_Float16*)p; p += nAct * 2;
  _Float16* wq16 = (_Float16*)p; p += nW * 2;     // transposed [n][k]
  _Float16* wk16 = (_Float16*)p; p += nW * 2;
  _Float16* wv16 = (_Float16*)p; p += nW * 2;
  _Float16* wo16 = (_Float16*)p; p += nW * 2;
  _Float16* qp   = (_Float16*)p; p += nAct * 2;
  _Float16* kp2  = (_Float16*)p; p += nAct * 2;
  _Float16* vT   = (_Float16*)p; p += nAct * 2;   // [b][h][d][s]
  _Float16* attn = qf;

  const int nAct4 = (int)(nAct / 4);
  cvt_f32_f16<<<nAct4 / 256, 256, 0, stream>>>((const float4*)query, (v4h*)qf, nAct4);
  cvt_f32_f16<<<nAct4 / 256, 256, 0, stream>>>((const float4*)key,   (v4h*)kf, nAct4);
  cvt_f32_f16<<<nAct4 / 256, 256, 0, stream>>>((const float4*)value, (v4h*)vf, nAct4);

  const dim3 wgrid(DIM / 64, DIM / 64);  // (16,16)
  cvt_w_transpose<<<wgrid, 256, 0, stream>>>(Wq, wq16);
  cvt_w_transpose<<<wgrid, 256, 0, stream>>>(Wk, wk16);
  cvt_w_transpose<<<wgrid, 256, 0, stream>>>(Wv, wv16);
  cvt_w_transpose<<<wgrid, 256, 0, stream>>>(Wo, wo16);

  const dim3 ggrid(DIM / 128, NROWS / 128);  // (8, 64)
  gemm_wmma<_Float16, true,  false><<<ggrid, 256, 0, stream>>>(qf, wq16, bq, qp,  DIM, DIM);
  gemm_wmma<_Float16, true,  false><<<ggrid, 256, 0, stream>>>(kf, wk16, bk, kp2, DIM, DIM);
  gemm_wmma<_Float16, false, true ><<<ggrid, 256, 0, stream>>>(vf, wv16, bv, vT,  DIM, DIM);

  attn_fwd<<<dim3(SEQ / 128, BATCH * NHEAD), 256, 0, stream>>>(qp, kp2, vT, attn);

  gemm_wmma<float, false, false><<<ggrid, 256, 0, stream>>>(attn, wo16, bo, out, DIM, DIM);
}
